// GatingPredictor_8478265442436
// MI455X (gfx1250) — compile-verified, matrix operand order checked by reading, not run on hardware
//
#include <hip/hip_runtime.h>
#include <hip/hip_bf16.h>
#include <math.h>

// ---- problem shape (fixed by reference) ----
#define BB 4
#define HH 8
#define SS 4096
#define DD 128
#define INNER 256     // D*EXPAND
#define IN_DIM 384    // 3*D
#define K_STEPS (IN_DIM / 32)   // 12
#define SPAD 392      // padded f16 row stride: 784B = 196 dwords -> 4-bank stagger
#define CHUNKS 32     // s-chunks per (b,h)
#define TILES_PER_WG 8          // 8 tiles * 16 tokens = 128 tokens per WG
#define TOK_PER_WG (TILES_PER_WG * 16)

typedef __attribute__((ext_vector_type(16))) _Float16 v16h;
typedef __attribute__((ext_vector_type(8)))  _Float16 v8h;
typedef __attribute__((ext_vector_type(8)))  float    v8f;

__global__ __launch_bounds__(256, 1)
void gating_predictor_kernel(const float* __restrict__ pre,
                             const float* __restrict__ post,
                             const float* __restrict__ val,
                             const float* __restrict__ nw_pre,
                             const float* __restrict__ nw_post,
                             const float* __restrict__ nw_v,
                             const float* __restrict__ w1,
                             const float* __restrict__ b1,
                             const float* __restrict__ w2,
                             const float* __restrict__ b2,
                             float* __restrict__ out) {
    // ---- LDS (CDNA5: 320KB per workgroup; this uses ~215KB) ----
    __shared__ _Float16 wlds[INNER * SPAD];   // 200704 B : whole head's w1 in f16
    __shared__ _Float16 slds[16 * SPAD];      //  12544 B : one 16-token state tile
    __shared__ float b1l[INNER];
    __shared__ float w2l[INNER];
    __shared__ float gacc[16];

    const int tid  = threadIdx.x;
    const int lane = tid & 31;
    const int wid  = tid >> 5;

    const int bh    = blockIdx.x / CHUNKS;     // 0..31  ([B,H] flat)
    const int chunk = blockIdx.x % CHUNKS;
    const int h     = bh % HH;
    const size_t abase  = (size_t)bh * SS * DD;
    const int    s0base = chunk * TOK_PER_WG;

    // ---- stage w1[h] -> LDS as f16 (coalesced float4 reads) ----
    {
        const float4* w1h = reinterpret_cast<const float4*>(w1 + (size_t)h * INNER * IN_DIM);
        const int n4 = INNER * IN_DIM / 4;     // 24576 float4
        #pragma unroll 4
        for (int idx = tid; idx < n4; idx += 256) {
            const int col = idx / (IN_DIM / 4);
            const int kk  = idx % (IN_DIM / 4);
            const float4 v = w1h[idx];
            _Float16* dst = &wlds[col * SPAD + kk * 4];
            dst[0] = (_Float16)v.x; dst[1] = (_Float16)v.y;
            dst[2] = (_Float16)v.z; dst[3] = (_Float16)v.w;
        }
        b1l[tid] = b1[h * INNER + tid];
        w2l[tid] = w2[h * INNER + tid];
    }
    const float b2h = b2[h];

    const float* const src3[3] = {pre + abase, post + abase, val + abase};
    const float* const nw3[3]  = {nw_pre, nw_post, nw_v};

    const int token = tid >> 4;          // 0..15 : token within s-tile
    const int l16   = tid & 15;          // 16 lanes cooperate per token
    const int arow  = lane & 15;         // A-matrix row (M)
    const int khalf = lane >> 4;         // K-half select per ISA A/B lane layout
    const int col0  = (wid * 2 + 0) * 16 + (lane & 15);
    const int col1  = col0 + 16;

    // ---- prologue: prefetch s-tile 0 activations into registers ----
    float4 cur[3][2];
    #pragma unroll
    for (int p = 0; p < 3; ++p) {
        const float* sp = src3[p] + (size_t)(s0base + token) * DD + l16 * 8;
        cur[p][0] = reinterpret_cast<const float4*>(sp)[0];
        cur[p][1] = reinterpret_cast<const float4*>(sp)[1];
    }
    __syncthreads();   // wlds/b1l/w2l ready

    for (int st = 0; st < TILES_PER_WG; ++st) {
        const int s0 = s0base + st * 16;

        if (tid < 16) gacc[tid] = 0.0f;

        // ---- RMSNorm (from prefetched regs) -> f16 states tile in LDS ----
        #pragma unroll
        for (int p = 0; p < 3; ++p) {
            float xs[8] = {cur[p][0].x, cur[p][0].y, cur[p][0].z, cur[p][0].w,
                           cur[p][1].x, cur[p][1].y, cur[p][1].z, cur[p][1].w};
            float ss = 0.0f;
            #pragma unroll
            for (int j = 0; j < 8; ++j) ss += xs[j] * xs[j];
            #pragma unroll
            for (int m = 8; m >= 1; m >>= 1) ss += __shfl_xor(ss, m, 16);
            const float inv = rsqrtf(ss * (1.0f / (float)DD) + 1e-6f);
            const float* nw = nw3[p] + l16 * 8;
            v8h hv;
            #pragma unroll
            for (int j = 0; j < 8; ++j) hv[j] = (_Float16)(xs[j] * inv * nw[j]);
            *reinterpret_cast<v8h*>(&slds[token * SPAD + p * DD + l16 * 8]) = hv;
        }
        __syncthreads();

        // ---- issue next tile's HBM loads now; split LOADcnt/DScnt means the
        //      WMMA loop below (DS-only) never waits on them ----
        float4 nxt[3][2];
        if (st + 1 < TILES_PER_WG) {
            #pragma unroll
            for (int p = 0; p < 3; ++p) {
                const float* sp = src3[p] + (size_t)(s0 + 16 + token) * DD + l16 * 8;
                nxt[p][0] = reinterpret_cast<const float4*>(sp)[0];
                nxt[p][1] = reinterpret_cast<const float4*>(sp)[1];
            }
        } else {
            #pragma unroll
            for (int p = 0; p < 3; ++p) {
                nxt[p][0] = make_float4(0.f, 0.f, 0.f, 0.f);
                nxt[p][1] = make_float4(0.f, 0.f, 0.f, 0.f);
            }
        }

        // ---- 16x16x32 f16 WMMA over K=384; each wave owns 2 N-tiles ----
        v8f acc0 = {}; v8f acc1 = {};
        #pragma unroll
        for (int ks = 0; ks < K_STEPS; ++ks) {
            const int ka = ks * 32 + khalf * 8;       // K {0..7 | 8..15} of step
            const int kc = ka + 16;                   // K {16..23 | 24..31}
            const v8h aLo = *reinterpret_cast<const v8h*>(&slds[arow * SPAD + ka]);
            const v8h aHi = *reinterpret_cast<const v8h*>(&slds[arow * SPAD + kc]);
            const v8h b0L = *reinterpret_cast<const v8h*>(&wlds[col0 * SPAD + ka]);
            const v8h b0H = *reinterpret_cast<const v8h*>(&wlds[col0 * SPAD + kc]);
            const v8h b1L = *reinterpret_cast<const v8h*>(&wlds[col1 * SPAD + ka]);
            const v8h b1H = *reinterpret_cast<const v8h*>(&wlds[col1 * SPAD + kc]);
            const v16h a   = __builtin_shufflevector(aLo, aHi, 0,1,2,3,4,5,6,7,8,9,10,11,12,13,14,15);
            const v16h bb0 = __builtin_shufflevector(b0L, b0H, 0,1,2,3,4,5,6,7,8,9,10,11,12,13,14,15);
            const v16h bb1 = __builtin_shufflevector(b1L, b1H, 0,1,2,3,4,5,6,7,8,9,10,11,12,13,14,15);
            acc0 = __builtin_amdgcn_wmma_f32_16x16x32_f16(false, a, false, bb0, (short)0, acc0, false, false);
            acc1 = __builtin_amdgcn_wmma_f32_16x16x32_f16(false, a, false, bb1, (short)0, acc1, false, false);
        }

        // ---- epilogue: +b1, SiLU, *w2, reduce over N ----
        const float bA = b1l[col0], bB = b1l[col1];
        const float wA = w2l[col0], wB = w2l[col1];
        float r[8];
        #pragma unroll
        for (int v = 0; v < 8; ++v) {
            const float hA = acc0[v] + bA;
            const float hB = acc1[v] + bB;
            const float sA = hA / (1.0f + __expf(-hA));   // silu
            const float sB = hB / (1.0f + __expf(-hB));
            r[v] = sA * wA + sB * wB;
        }
        #pragma unroll
        for (int m = 8; m >= 1; m >>= 1) {
            #pragma unroll
            for (int v = 0; v < 8; ++v) r[v] += __shfl_xor(r[v], m, 16);
        }
        if ((lane & 15) == 0) {
            const int rbase = (lane >> 4) * 8;   // lanes 0-15 -> rows 0..7; 16-31 -> rows 8..15
            #pragma unroll
            for (int v = 0; v < 8; ++v) atomicAdd(&gacc[rbase + v], r[v]);
        }
        __syncthreads();

        if (tid < 16) {
            const float g = gacc[tid] + b2h;
            out[(size_t)bh * SS + s0 + tid] = 1.0f / (1.0f + __expf(-g));
        }
        __syncthreads();   // protect gacc/slds before next tile

        // rotate prefetch registers
        #pragma unroll
        for (int p = 0; p < 3; ++p) { cur[p][0] = nxt[p][0]; cur[p][1] = nxt[p][1]; }
    }
}

extern "C" void kernel_launch(void* const* d_in, const int* in_sizes, int n_in,
                              void* d_out, int out_size, void* d_ws, size_t ws_size,
                              hipStream_t stream) {
    (void)in_sizes; (void)n_in; (void)out_size; (void)d_ws; (void)ws_size;
    const float* pre     = (const float*)d_in[0];
    const float* post    = (const float*)d_in[1];
    const float* val     = (const float*)d_in[2];
    const float* nw_pre  = (const float*)d_in[3];
    const float* nw_post = (const float*)d_in[4];
    const float* nw_v    = (const float*)d_in[5];
    const float* w1      = (const float*)d_in[6];
    const float* b1      = (const float*)d_in[7];
    const float* w2      = (const float*)d_in[8];
    const float* b2      = (const float*)d_in[9];
    float* out = (float*)d_out;

    dim3 grid(BB * HH * CHUNKS);   // 1024 workgroups
    dim3 block(256);               // 8 wave32 waves
    gating_predictor_kernel<<<grid, block, 0, stream>>>(
        pre, post, val, nw_pre, nw_post, nw_v, w1, b1, w2, b2, out);
}
